// PaiAutoencoder_83391085019893
// MI455X (gfx1250) — matrix operand order repaired and checked by reading, not compiled, not run on hardware
//
#include <hip/hip_runtime.h>
#include <math.h>

// ---------------------------------------------------------------------------
// MI455X (gfx1250) implementation of the PaiConv mesh autoencoder.
// All dense contractions (pai_conv GEMMs, adjacency GEMMs, FC layers) go
// through v_wmma_f32_16x16x32_bf16 (wave32, 16x16 C-tile per wave).
// K is padded to multiples of 32 and B is pre-swizzled into fragment layout
// so the GEMM inner loop is branch-free bulk b128 loads + wmma.
// ---------------------------------------------------------------------------

typedef __bf16 bf16;
typedef __attribute__((ext_vector_type(16))) __bf16 v16bf;
typedef __attribute__((ext_vector_type(8)))  __bf16 v8bf;
typedef __attribute__((ext_vector_type(8)))  float  v8f;

#define TPB 256

// ---------------------------- small kernels --------------------------------

__global__ void mean3_kernel(const float* __restrict__ v, int N,
                             float* __restrict__ mean) {
  __shared__ float s0[TPB], s1[TPB], s2[TPB];
  int t = threadIdx.x;
  float a0 = 0.f, a1 = 0.f, a2 = 0.f;
  for (int n = t; n < N; n += TPB) {
    a0 += v[n * 3 + 0];
    a1 += v[n * 3 + 1];
    a2 += v[n * 3 + 2];
  }
  s0[t] = a0; s1[t] = a1; s2[t] = a2;
  __syncthreads();
  for (int st = TPB / 2; st > 0; st >>= 1) {
    if (t < st) { s0[t] += s0[t + st]; s1[t] += s1[t + st]; s2[t] += s2[t + st]; }
    __syncthreads();
  }
  if (t == 0) {
    mean[0] = s0[0] / (float)N;
    mean[1] = s1[0] / (float)N;
    mean[2] = s2[0] / (float)N;
  }
}

__global__ void make_nv_kernel(const float* __restrict__ v,
                               const float* __restrict__ mean,
                               float* __restrict__ nv, int n3) {
  int i = blockIdx.x * blockDim.x + threadIdx.x;
  if (i >= n3) return;
  float m = mean[i % 3];
  nv[i] = 0.1f * (v[i] - m) / m;
}

// out[n,d] = sum_j v[n,j] * W[j,d], W is (3,32)
__global__ void proj32_kernel(const float* __restrict__ v,
                              const float* __restrict__ W,
                              float* __restrict__ out, int N) {
  int i = blockIdx.x * blockDim.x + threadIdx.x;
  if (i >= N * 32) return;
  int d = i & 31, n = i >> 5;
  float s = 0.f;
#pragma unroll
  for (int j = 0; j < 3; ++j) s += v[n * 3 + j] * W[j * 32 + d];
  out[i] = s;
}

// One block per query row: scores -> iterative top-k -> prior-blended weights.
__global__ void attpool_weights_kernel(const float* __restrict__ qp,
                                       const float* __restrict__ kp,
                                       const float* __restrict__ prior,
                                       int Nq, int Nk, int topk,
                                       int* __restrict__ idx_out,
                                       float* __restrict__ w_out) {
  __shared__ float sc[5024];
  __shared__ float redv[TPB];
  __shared__ int   redi[TPB];
  __shared__ float qv[32];
  __shared__ float topv[32];
  __shared__ int   topi[32];
  int q = blockIdx.x;
  int t = threadIdx.x;
  if (t < 32) qv[t] = qp[q * 32 + t];
  __syncthreads();
  const float scale = 0.17677669529663687f;  // 1/sqrt(32)
  for (int n = t; n < Nk; n += TPB) {
    float s = 0.f;
#pragma unroll
    for (int d = 0; d < 32; ++d) s += qv[d] * kp[n * 32 + d];
    sc[n] = s * scale;
  }
  __syncthreads();
  for (int sel = 0; sel < topk; ++sel) {
    float best = -3.0e38f;
    int bi = 0x7fffffff;
    for (int n = t; n < Nk; n += TPB) {
      float v = sc[n];
      if (v > best) { best = v; bi = n; }
    }
    redv[t] = best; redi[t] = bi;
    __syncthreads();
    for (int st = TPB / 2; st > 0; st >>= 1) {
      if (t < st) {
        float v2 = redv[t + st]; int i2 = redi[t + st];
        if (v2 > redv[t] || (v2 == redv[t] && i2 < redi[t])) {
          redv[t] = v2; redi[t] = i2;
        }
      }
      __syncthreads();
    }
    if (t == 0) {
      topv[sel] = redv[0];
      topi[sel] = redi[0];
      sc[redi[0]] = -3.0e38f;
    }
    __syncthreads();
  }
  if (t == 0) {
    float mx = topv[0];
    float sum = 0.f;
    for (int s2 = 0; s2 < topk; ++s2) sum += expf(topv[s2] - mx);
    for (int s2 = 0; s2 < topk; ++s2) {
      float sm = expf(topv[s2] - mx) / sum;
      w_out[q * topk + s2] =
          0.8f * prior[(size_t)q * Nk + topi[s2]] + 0.2f * sm;
      idx_out[q * topk + s2] = topi[s2];
    }
  }
}

// out[b,q,f] = sum_t w[q,t] * x[b, idx[q,t], f]
__global__ void attpool_apply_kernel(const float* __restrict__ x,
                                     const int* __restrict__ idx,
                                     const float* __restrict__ w,
                                     float* __restrict__ out, int Bv, int Nq,
                                     int Nk, int F, int topk) {
  long i = (long)blockIdx.x * blockDim.x + threadIdx.x;
  long total = (long)Bv * Nq * F;
  if (i >= total) return;
  int f = (int)(i % F);
  long r = i / F;
  int q = (int)(r % Nq);
  int b = (int)(r / Nq);
  float s = 0.f;
  for (int t = 0; t < topk; ++t) {
    int src = idx[q * topk + t];
    s += w[q * topk + t] * x[((long)b * Nk + src) * F + f];
  }
  out[i] = s;
}

// f[n, 0..23] = neighbor diffs, f[n, 24..26] = v[n]
__global__ void fourier_f_kernel(const float* __restrict__ v,
                                 const int* __restrict__ S,
                                 float* __restrict__ f, int N) {
  int n = blockIdx.x * blockDim.x + threadIdx.x;
  if (n >= N) return;
  float vx = v[n * 3 + 0], vy = v[n * 3 + 1], vz = v[n * 3 + 2];
  for (int k = 1; k < 9; ++k) {
    int s = S[n * 9 + k];
    f[n * 27 + (k - 1) * 3 + 0] = v[s * 3 + 0] - vx;
    f[n * 27 + (k - 1) * 3 + 1] = v[s * 3 + 1] - vy;
    f[n * 27 + (k - 1) * 3 + 2] = v[s * 3 + 2] - vz;
  }
  f[n * 27 + 24] = vx;
  f[n * 27 + 25] = vy;
  f[n * 27 + 26] = vz;
}

// one block per feature column (27 columns)
__global__ void colminmax_kernel(const float* __restrict__ f, int N,
                                 float* __restrict__ lo,
                                 float* __restrict__ hi) {
  __shared__ float smin[TPB], smax[TPB];
  int j = blockIdx.x;
  int t = threadIdx.x;
  float mn = 3.0e38f, mx = -3.0e38f;
  for (int n = t; n < N; n += TPB) {
    float v = f[n * 27 + j];
    mn = fminf(mn, v);
    mx = fmaxf(mx, v);
  }
  smin[t] = mn; smax[t] = mx;
  __syncthreads();
  for (int st = TPB / 2; st > 0; st >>= 1) {
    if (t < st) {
      smin[t] = fminf(smin[t], smin[t + st]);
      smax[t] = fmaxf(smax[t], smax[t + st]);
    }
    __syncthreads();
  }
  if (t == 0) { lo[j] = smin[0]; hi[j] = smax[0]; }
}

// tv[n, j] = sin(fo[j]) (j<64), tv[n, 64+j] = cos(fo[j]); stored bf16
__global__ void fourier_tv_kernel(const float* __restrict__ f,
                                  const float* __restrict__ lo,
                                  const float* __restrict__ hi,
                                  const float* __restrict__ B1,
                                  const float* __restrict__ B2,
                                  bf16* __restrict__ tv, int N) {
  int i = blockIdx.x * blockDim.x + threadIdx.x;
  if (i >= N * 64) return;
  int j = i & 63, n = i >> 6;
  float s = 0.f;
  if (j < 32) {
    for (int p = 0; p < 24; ++p) {
      float tt = ((f[n * 27 + p] - lo[p]) / (hi[p] - lo[p]) - 0.5f) * 5.0f;
      s += tt * B1[p * 32 + j];
    }
  } else {
    int jj = j - 32;
#pragma unroll
    for (int p = 0; p < 3; ++p) {
      float tt =
          ((f[n * 27 + 24 + p] - lo[24 + p]) / (hi[24 + p] - lo[24 + p]) -
           0.5f) * 5.0f;
      s += tt * B2[p * 32 + jj];
    }
  }
  s *= 6.283185307179586f;
  tv[(long)n * 128 + j] = (bf16)sinf(s);
  tv[(long)n * 128 + 64 + j] = (bf16)cosf(s);
}

__global__ void f32_to_bf16_kernel(const float* __restrict__ a,
                                   bf16* __restrict__ b, long n) {
  long i = (long)blockIdx.x * blockDim.x + threadIdx.x;
  if (i < n) b[i] = (bf16)a[i];
}

// in-place softmax over groups of 9 contiguous values: a[(n*9+k)*9 + t]
__global__ void softmax9_kernel(float* __restrict__ a, int N) {
  int i = blockIdx.x * blockDim.x + threadIdx.x;
  if (i >= N * 9) return;
  float* p = a + (long)i * 9;
  float mx = p[0];
#pragma unroll
  for (int t = 1; t < 9; ++t) mx = fmaxf(mx, p[t]);
  float e[9], sum = 0.f;
#pragma unroll
  for (int t = 0; t < 9; ++t) { e[t] = expf(p[t] - mx); sum += e[t]; }
  float inv = 1.0f / sum;
#pragma unroll
  for (int t = 0; t < 9; ++t) p[t] = e[t] * inv;
}

// xm[b,n,:] row of length Kp (zero-padded past 9*C):
// xm[b,n,t*C+c] = sum_k x[b, S[n,k], c] * adj[n, k, t]   (bf16 output)
__global__ void gather_mix_kernel(const float* __restrict__ x,
                                  const int* __restrict__ S,
                                  const float* __restrict__ adj,
                                  bf16* __restrict__ xm, int Bv, int N, int C,
                                  int Kp) {
  long i = (long)blockIdx.x * blockDim.x + threadIdx.x;
  long total = (long)Bv * N * Kp;
  if (i >= total) return;
  int kidx = (int)(i % Kp);
  long r = i / Kp;
  int n = (int)(r % N);
  int b = (int)(r / N);
  if (kidx >= 9 * C) { xm[i] = (bf16)0.0f; return; }
  int t = kidx / C;
  int c = kidx - t * C;
  const float* adjr = adj + (long)n * 81;
  float s = 0.f;
#pragma unroll
  for (int k = 0; k < 9; ++k) {
    int src = S[n * 9 + k];
    s += x[((long)b * N + src) * C + c] * adjr[k * 9 + t];
  }
  xm[i] = (bf16)s;
}

// Swizzle W (K,N) f32 row-major into per-wave WMMA B fragments (bf16),
// zero-padding K to Kp=ceil32 and N to Ntiles*16.
// out[((kt*Ntiles + tn)*32 + lane)*16 + e] =
//   W[kt*32 + (lane>>4)*8 + (e&7) + (e>>3)*16, tn*16 + (lane&15)]
__global__ void swizzle_w_kernel(const float* __restrict__ W,
                                 bf16* __restrict__ out, int K, int N,
                                 int Ntiles, long total) {
  long i = (long)blockIdx.x * blockDim.x + threadIdx.x;
  if (i >= total) return;
  int e = (int)(i & 15);
  int lane = (int)((i >> 4) & 31);
  long r = i >> 9;
  int tn = (int)(r % Ntiles);
  int kt = (int)(r / Ntiles);
  int l = lane & 15, half = lane >> 4;
  int k = kt * 32 + half * 8 + (e & 7) + ((e >> 3) << 4);
  int n = tn * 16 + l;
  out[i] = (k < K && n < N) ? (bf16)W[(long)k * N + n] : (bf16)0.0f;
}

// ------------------------------ WMMA GEMM ----------------------------------
// C[M,N] = A[M,Kp] @ Bsw + bias, optional ELU.
// A: bf16 row-major, row stride Kp (multiple of 32, zero-padded tail).
// Bsw: pre-swizzled fragments (see swizzle_w_kernel). C: f32 row-major (N).
// One wave per 16x16 C tile; inner loop = 2x b128 (A) + 2x b128 (B) + wmma.
__global__ void __launch_bounds__(32)
wmma_gemm_kernel(const bf16* __restrict__ A, const bf16* __restrict__ Bsw,
                 const float* __restrict__ bias, float* __restrict__ C, int M,
                 int N, int Kp, int act) {
  const int lane = threadIdx.x;
  const int l = lane & 15;
  const int half = lane >> 4;
  const int tn = blockIdx.x;
  const int tm = blockIdx.y;

  int arow = tm * 16 + l;
  if (arow >= M) arow = M - 1;  // clamped dup row; store is masked
  const bf16* Ap = A + (size_t)arow * Kp + (half ? 8 : 0);
  const bf16* Bp = Bsw + (size_t)tn * 512 + (size_t)lane * 16;
  const size_t bstep = (size_t)gridDim.x * 512;

  v8f acc = {};
  for (int kk = 0; kk < Kp; kk += 32) {
    __builtin_prefetch(Ap + kk + 32, 0, 3);
    // A 16x32 bf16 fragment: lanes 0-15 rows hold K {0..7,16..23},
    // lanes 16-31 hold K {8..15,24..31} -> two contiguous 16B chunks.
    v8bf a0 = *(const v8bf*)(Ap + kk);
    v8bf a1 = *(const v8bf*)(Ap + kk + 16);
    v16bf af = __builtin_shufflevector(a0, a1, 0, 1, 2, 3, 4, 5, 6, 7, 8, 9,
                                       10, 11, 12, 13, 14, 15);
    v16bf bfv = *(const v16bf*)Bp;  // pre-swizzled: fully contiguous
    Bp += bstep;
    acc = __builtin_amdgcn_wmma_f32_16x16x32_bf16(false, af, false, bfv,
                                                  (short)0, acc, false, false);
  }
  // C/D layout: VGPR j -> row (half*8 + j), lane%16 -> column.
  int cn = tn * 16 + l;
  if (cn < N) {
    float bv = bias ? bias[cn] : 0.0f;
#pragma unroll
    for (int j = 0; j < 8; ++j) {
      int cm = tm * 16 + half * 8 + j;
      if (cm < M) {
        float v = acc[j] + bv;
        if (act) v = (v > 0.0f) ? v : expm1f(v);
        C[(long)cm * N + cn] = v;
      }
    }
  }
}

// ------------------------------ host side ----------------------------------

static inline int cdiv_l(long a, long b) { return (int)((a + b - 1) / b); }
static inline int rup32(int k) { return (k + 31) & ~31; }

static inline void launch_gemm(const bf16* A, const bf16* Bsw,
                               const float* bias, float* C, int M, int N,
                               int Kp, int act, hipStream_t s) {
  dim3 grid((N + 15) / 16, (M + 15) / 16);
  wmma_gemm_kernel<<<grid, 32, 0, s>>>(A, Bsw, bias, C, M, N, Kp, act);
}

static inline void launch_swizzle(const float* W, bf16* out, int K, int N,
                                  hipStream_t s) {
  int Ntiles = (N + 15) / 16;
  int Kp = rup32(K);
  long total = (long)Kp * Ntiles * 16;
  swizzle_w_kernel<<<cdiv_l(total, TPB), TPB, 0, s>>>(W, out, K, N, Ntiles,
                                                      total);
}

extern "C" void kernel_launch(void* const* d_in, const int* in_sizes, int n_in,
                              void* d_out, int out_size, void* d_ws,
                              size_t ws_size, hipStream_t stream) {
  (void)in_sizes; (void)n_in; (void)out_size; (void)ws_size;
  static const int NSv[4] = {5024, 1257, 315, 80};
  static const int FEv[4] = {3, 64, 128, 256};
  static const int FDv[5] = {256, 128, 64, 64, 3};
  const int Bv = 16;

  // ---- inputs (setup_inputs dict order, nested lists flattened) ----
  const float* x_in = (const float*)d_in[0];
  const float* verts[4];
  for (int i = 0; i < 4; ++i) verts[i] = (const float*)d_in[1 + i];
  const float* B1 = (const float*)d_in[5];
  const float* B2 = (const float*)d_in[6];
  const float* Dp[3];
  for (int i = 0; i < 3; ++i) Dp[i] = (const float*)d_in[7 + i];
  const float* Up[3];
  for (int i = 0; i < 3; ++i) Up[i] = (const float*)d_in[10 + i];
  const float *e_mW[3], *e_mb[3], *e_cW[3], *e_cb[3];
  for (int i = 0; i < 3; ++i) {
    e_mW[i] = (const float*)d_in[13 + 4 * i];
    e_mb[i] = (const float*)d_in[14 + 4 * i];
    e_cW[i] = (const float*)d_in[15 + 4 * i];
    e_cb[i] = (const float*)d_in[16 + 4 * i];
  }
  const float *dm_mW[4], *dm_mb[4], *dm_cW[4], *dm_cb[4];
  for (int i = 0; i < 4; ++i) {
    dm_mW[i] = (const float*)d_in[25 + 4 * i];
    dm_mb[i] = (const float*)d_in[26 + 4 * i];
    dm_cW[i] = (const float*)d_in[27 + 4 * i];
    dm_cb[i] = (const float*)d_in[28 + 4 * i];
  }
  const float *aeWq[3], *aeWk[3], *adWq[3], *adWk[3];
  for (int i = 0; i < 3; ++i) {
    aeWq[i] = (const float*)d_in[41 + 2 * i];
    aeWk[i] = (const float*)d_in[42 + 2 * i];
    adWq[i] = (const float*)d_in[47 + 2 * i];
    adWk[i] = (const float*)d_in[48 + 2 * i];
  }
  const float* fcEW = (const float*)d_in[53];
  const float* fcEb = (const float*)d_in[54];
  const float* fcDW = (const float*)d_in[55];
  const float* fcDb = (const float*)d_in[56];
  const int* Sg[4];
  for (int i = 0; i < 4; ++i) Sg[i] = (const int*)d_in[57 + i];

  // ---- workspace carve (bump allocator, 256B aligned) ----
  char* ws = (char*)d_ws;
  size_t off = 0;
  auto alloc = [&](size_t bytes) -> void* {
    void* p = ws + off;
    off += (bytes + 255) & ~(size_t)255;
    return p;
  };
  float* meanb = (float*)alloc(3 * 4);
  float* nv[4];
  for (int i = 0; i < 4; ++i) nv[i] = (float*)alloc((size_t)NSv[i] * 3 * 4);
  float* obuf[4];
  obuf[0] = nullptr;
  for (int i = 1; i < 4; ++i) obuf[i] = (float*)alloc((size_t)NSv[i] * 3 * 4);
  bf16* tv[4];
  for (int i = 0; i < 4; ++i) tv[i] = (bf16*)alloc((size_t)NSv[i] * 128 * 2);
  float* fbuf = (float*)alloc((size_t)5024 * 27 * 4);
  float* lo = (float*)alloc(27 * 4);
  float* hi = (float*)alloc(27 * 4);
  float* qp = (float*)alloc((size_t)5024 * 32 * 4);
  float* kp = (float*)alloc((size_t)5024 * 32 * 4);
  int* idxE[3]; float* wE[3];
  for (int i = 0; i < 3; ++i) {
    idxE[i] = (int*)alloc((size_t)NSv[i + 1] * 8 * 4);
    wE[i] = (float*)alloc((size_t)NSv[i + 1] * 8 * 4);
  }
  int* idxD[3]; float* wD[3];
  for (int j = 0; j < 3; ++j) {
    idxD[j] = (int*)alloc((size_t)NSv[j] * 32 * 4);
    wD[j] = (float*)alloc((size_t)NSv[j] * 32 * 4);
  }
  float* adj = (float*)alloc((size_t)5024 * 81 * 4);
  float* hA = (float*)alloc((size_t)16 * 5024 * 64 * 4);
  float* hB = (float*)alloc((size_t)16 * 5024 * 64 * 4);
  bf16* xm = (bf16*)alloc((size_t)16 * 5024 * 576 * 2);
  bf16* wbuf = (bf16*)alloc((size_t)20480 * 128 * 2);
  bf16* abuf = (bf16*)alloc((size_t)16 * 20480 * 2);

  // ---- 1. vertex normalization ----
  mean3_kernel<<<1, TPB, 0, stream>>>(verts[0], NSv[0], meanb);
  for (int i = 0; i < 4; ++i)
    make_nv_kernel<<<cdiv_l((long)NSv[i] * 3, TPB), TPB, 0, stream>>>(
        verts[i], meanb, nv[i], NSv[i] * 3);

  // ---- 2. encoder pooling weights + coarse vertex chain o[] ----
  const float* ocur[4];
  ocur[0] = verts[0];
  for (int i = 0; i < 3; ++i) {
    proj32_kernel<<<cdiv_l((long)NSv[i + 1] * 32, TPB), TPB, 0, stream>>>(
        nv[i + 1], aeWq[i], qp, NSv[i + 1]);
    proj32_kernel<<<cdiv_l((long)NSv[i] * 32, TPB), TPB, 0, stream>>>(
        nv[i], aeWk[i], kp, NSv[i]);
    attpool_weights_kernel<<<NSv[i + 1], TPB, 0, stream>>>(
        qp, kp, Dp[i], NSv[i + 1], NSv[i], 8, idxE[i], wE[i]);
    attpool_apply_kernel<<<cdiv_l((long)NSv[i + 1] * 3, TPB), TPB, 0, stream>>>(
        ocur[i], idxE[i], wE[i], obuf[i + 1], 1, NSv[i + 1], NSv[i], 3, 8);
    ocur[i + 1] = obuf[i + 1];
  }

  // ---- 3. Fourier features tv[i] (bf16, (N,128)) ----
  for (int i = 0; i < 4; ++i) {
    fourier_f_kernel<<<cdiv_l(NSv[i], TPB), TPB, 0, stream>>>(ocur[i], Sg[i],
                                                              fbuf, NSv[i]);
    colminmax_kernel<<<27, TPB, 0, stream>>>(fbuf, NSv[i], lo, hi);
    fourier_tv_kernel<<<cdiv_l((long)NSv[i] * 64, TPB), TPB, 0, stream>>>(
        fbuf, lo, hi, B1, B2, tv[i], NSv[i]);
  }

  // ---- 4. decoder pooling weights ----
  for (int j = 0; j < 3; ++j) {
    proj32_kernel<<<cdiv_l((long)NSv[j] * 32, TPB), TPB, 0, stream>>>(
        nv[j], adWq[j], qp, NSv[j]);
    proj32_kernel<<<cdiv_l((long)NSv[j + 1] * 32, TPB), TPB, 0, stream>>>(
        nv[j + 1], adWk[j], kp, NSv[j + 1]);
    attpool_weights_kernel<<<NSv[j], TPB, 0, stream>>>(
        qp, kp, Up[j], NSv[j], NSv[j + 1], 32, idxD[j], wD[j]);
  }

  // ---- 5. encoder convs + pools ----
  const float* cur = x_in;
  for (int i = 0; i < 3; ++i) {
    int N = NSv[i], Cin = FEv[i], Cout = FEv[i + 1];
    int Kc = 9 * Cin, Kcp = rup32(Kc);
    // adjacency: softmax((tv @ mW + mb).reshape(N,9,9))
    launch_swizzle(e_mW[i], wbuf, 128, 81, stream);
    launch_gemm(tv[i], wbuf, e_mb[i], adj, N, 81, 128, 0, stream);
    softmax9_kernel<<<cdiv_l((long)N * 9, TPB), TPB, 0, stream>>>(adj, N);
    // gather + 9x9 mix -> (B*N, Kcp) bf16, zero tail
    gather_mix_kernel<<<cdiv_l((long)Bv * N * Kcp, TPB), TPB, 0, stream>>>(
        cur, Sg[i], adj, xm, Bv, N, Cin, Kcp);
    launch_swizzle(e_cW[i], wbuf, Kc, Cout, stream);
    launch_gemm(xm, wbuf, e_cb[i], hA, Bv * N, Cout, Kcp, 1, stream);
    attpool_apply_kernel<<<cdiv_l((long)Bv * NSv[i + 1] * Cout, TPB), TPB, 0,
                           stream>>>(hA, idxE[i], wE[i], hB, Bv, NSv[i + 1], N,
                                     Cout, 8);
    cur = hB;
  }

  // ---- 6. FC bottleneck: z = h@We+be ; h = z@Wd+bd ----
  f32_to_bf16_kernel<<<cdiv_l((long)Bv * 20480, TPB), TPB, 0, stream>>>(
      hB, abuf, (long)Bv * 20480);
  launch_swizzle(fcEW, wbuf, 20480, 128, stream);
  launch_gemm(abuf, wbuf, fcEb, hA, 16, 128, 20480, 0, stream);  // z in hA
  f32_to_bf16_kernel<<<cdiv_l(16 * 128, TPB), TPB, 0, stream>>>(hA, abuf,
                                                                16 * 128);
  launch_swizzle(fcDW, wbuf, 128, 20480, stream);
  launch_gemm(abuf, wbuf, fcDb, hB, 16, 20480, 128, 0, stream);  // (16,80,256)

  // ---- 7. decoder: upsample-pool + conv, 3 levels ----
  int Cin = 256;
  for (int i = 0; i < 3; ++i) {
    int j = 2 - i, N = NSv[j], Cout = FDv[i + 1];
    int Kc = 9 * Cin, Kcp = rup32(Kc);
    attpool_apply_kernel<<<cdiv_l((long)Bv * N * Cin, TPB), TPB, 0, stream>>>(
        hB, idxD[j], wD[j], hA, Bv, N, NSv[j + 1], Cin, 32);
    launch_swizzle(dm_mW[i], wbuf, 128, 81, stream);
    launch_gemm(tv[j], wbuf, dm_mb[i], adj, N, 81, 128, 0, stream);
    softmax9_kernel<<<cdiv_l((long)N * 9, TPB), TPB, 0, stream>>>(adj, N);
    gather_mix_kernel<<<cdiv_l((long)Bv * N * Kcp, TPB), TPB, 0, stream>>>(
        hA, Sg[j], adj, xm, Bv, N, Cin, Kcp);
    launch_swizzle(dm_cW[i], wbuf, Kc, Cout, stream);
    launch_gemm(xm, wbuf, dm_cb[i], hB, Bv * N, Cout, Kcp, 1, stream);
    Cin = Cout;
  }

  // ---- 8. final conv (level 0, no activation) -> d_out ----
  launch_swizzle(dm_mW[3], wbuf, 128, 81, stream);
  launch_gemm(tv[0], wbuf, dm_mb[3], adj, 5024, 81, 128, 0, stream);
  softmax9_kernel<<<cdiv_l((long)5024 * 9, TPB), TPB, 0, stream>>>(adj, 5024);
  gather_mix_kernel<<<cdiv_l((long)Bv * 5024 * 576, TPB), TPB, 0, stream>>>(
      hB, Sg[0], adj, xm, Bv, 5024, 64, 576);
  launch_swizzle(dm_cW[3], wbuf, 576, 3, stream);
  launch_gemm(xm, wbuf, dm_cb[3], (float*)d_out, Bv * 5024, 3, 576, 0, stream);
}
